// MultiHead_11287174054380
// MI455X (gfx1250) — compile-verified, matrix operand order checked by reading, not run on hardware
//
#include <hip/hip_runtime.h>

// ---------------------------------------------------------------------------
// MHA forward for MI455X (gfx1250, wave32, WMMA + double-buffered async-LDS).
// B=2, S=2048, D=768, H=12, DK=64.
// ---------------------------------------------------------------------------

#define Bv   2
#define Sv   2048
#define Dv   768
#define Hv   12
#define DKv  64

#define AS1 __attribute__((address_space(1)))
#define AS3 __attribute__((address_space(3)))

typedef __attribute__((ext_vector_type(16))) _Float16 v16h;
typedef __attribute__((ext_vector_type(8)))  _Float16 v8h;
typedef __attribute__((ext_vector_type(4)))  _Float16 v4h;
typedef __attribute__((ext_vector_type(8)))  float    v8f;
typedef int v4i __attribute__((vector_size(16)));   // matches builtin proto

#if defined(__has_builtin)
#  if __has_builtin(__builtin_amdgcn_global_load_async_to_lds_b128) && \
      __has_builtin(__builtin_amdgcn_s_wait_asynccnt)
#    define USE_ASYNC_LDS 1
#  endif
#endif
#ifndef USE_ASYNC_LDS
#  define USE_ASYNC_LDS 0
#endif

static __device__ inline v8f wmma_f16(v16h a, v16h b, v8f c) {
  // D = A(16x32 f16) * B(32x16 f16) + C(16x16 f32)
  return __builtin_amdgcn_wmma_f32_16x16x32_f16(
      /*neg_a=*/false, a, /*neg_b=*/false, b,
      /*c_mod=*/(short)0, c, /*reuse_a=*/false, /*reuse_b=*/false);
}

// CDNA5 16-bit WMMA operand layout (A and B mirror each other):
//   lane L: index-in-16 = L & 15 (M for A, N for B)
//   kbase = (L >> 4) * 8
//   fragment halves 0..7  -> K = kbase + 0..7
//   fragment halves 8..15 -> K = kbase + 16..23
// `rowk` points at element K=0 of this lane's row (contiguous along K).

static __device__ inline v16h frag_from_f16row(const _Float16* rowk, int lane) {
  const int kbase = (lane >> 4) << 3;
  v8h lo = *(const v8h*)(rowk + kbase);
  v8h hi = *(const v8h*)(rowk + kbase + 16);
  v16h f;
#pragma unroll
  for (int i = 0; i < 8; ++i) { f[i] = lo[i]; f[i + 8] = hi[i]; }
  return f;
}

static __device__ inline v16h frag_from_f16row_scaled(const _Float16* rowk, int lane,
                                                      _Float16 scale) {
  const int kbase = (lane >> 4) << 3;
  v8h lo = *(const v8h*)(rowk + kbase);
  v8h hi = *(const v8h*)(rowk + kbase + 16);
  v16h f;
#pragma unroll
  for (int i = 0; i < 8; ++i) { f[i] = lo[i] * scale; f[i + 8] = hi[i] * scale; }
  return f;
}

// Stage 2048 f16 elements (one 32x64 tile) from global to LDS with the whole
// wave; CDNA5 path uses GLOBAL_LOAD_ASYNC_TO_LDS_B128 (ASYNCcnt).  8 issues,
// each moving 32 lanes x 16 B = 512 B.
static __device__ inline void stage_tile_to_lds(const _Float16* __restrict__ g,
                                                _Float16* lds, int lane) {
#pragma unroll
  for (int s = 0; s < 8; ++s) {
    const _Float16* gp = g + s * 256 + lane * 8;
    _Float16* lp = lds + s * 256 + lane * 8;
#if USE_ASYNC_LDS
    __builtin_amdgcn_global_load_async_to_lds_b128((AS1 v4i*)gp, (AS3 v4i*)lp,
                                                   0, 0);
#else
    *(v8h*)lp = *(const v8h*)gp;
#endif
  }
}

// ---------------------------------------------------------------------------
// Kernel 0: one-time fp32 -> f16 downconvert (weights + activations); GEMM
// inner loops become pure f16 (no v_cvt), halving operand-fetch bytes.
// ---------------------------------------------------------------------------
__global__ void __launch_bounds__(256)
cvt_f32_to_f16_kernel(const float* __restrict__ src, _Float16* __restrict__ dst,
                      int n4) {
  const int i = blockIdx.x * 256 + threadIdx.x;
  if (i < n4) {
    float4 x = ((const float4*)src)[i];
    v4h h;
    h[0] = (_Float16)x.x; h[1] = (_Float16)x.y;
    h[2] = (_Float16)x.z; h[3] = (_Float16)x.w;
    ((v4h*)dst)[i] = h;
  }
}

// ---------------------------------------------------------------------------
// Kernel 1: fused Q/K/V projection.  y = x @ W^T + b, stored f16 as [B,H,S,DK].
// One wave computes a 32x64 output tile: 8 WMMAs per k-step, B frags reused 2x.
// ---------------------------------------------------------------------------
__global__ void __launch_bounds__(256)
qkv_proj_kernel(const _Float16* __restrict__ Xq, const _Float16* __restrict__ Xk,
                const _Float16* __restrict__ Xv,
                const _Float16* __restrict__ WQ, const float* __restrict__ bQ,
                const _Float16* __restrict__ WK, const float* __restrict__ bK,
                const _Float16* __restrict__ WV, const float* __restrict__ bV,
                _Float16* __restrict__ Qh, _Float16* __restrict__ Kh,
                _Float16* __restrict__ Vh) {
  const int lane = threadIdx.x & 31;
  const int gw   = blockIdx.x * 8 + (threadIdx.x >> 5);
  const int MT   = (Bv * Sv) / 32;   // 128 row tiles (32 rows each)
  const int NT   = Dv / 64;          // 12 column super-tiles
  const int mat  = gw / (MT * NT);   // 0:Q 1:K 2:V
  const int rem  = gw % (MT * NT);
  const int mtile = rem / NT;
  const int n64   = rem % NT;

  const _Float16*  X    = (mat == 0) ? Xq : (mat == 1) ? Xk : Xv;
  const _Float16*  W    = (mat == 0) ? WQ : (mat == 1) ? WK : WV;
  const float*     bias = (mat == 0) ? bQ : (mat == 1) ? bK : bV;
  _Float16*        Yh   = (mat == 0) ? Qh : (mat == 1) ? Kh : Vh;

  const int col = lane & 15;
  const int m0  = mtile * 32 + col;       // A rows for this lane (two halves)
  const int m1  = m0 + 16;

  const v8f z = {0.f, 0.f, 0.f, 0.f, 0.f, 0.f, 0.f, 0.f};
  v8f acc[2][4] = {{z, z, z, z}, {z, z, z, z}};

  for (int k0 = 0; k0 < Dv; k0 += 32) {
    v16h a0 = frag_from_f16row(X + (size_t)m0 * Dv + k0, lane);
    v16h a1 = frag_from_f16row(X + (size_t)m1 * Dv + k0, lane);
    if (k0 + 64 < Dv)
      __builtin_prefetch(X + (size_t)m0 * Dv + k0 + 64, 0, 1);  // global_prefetch
#pragma unroll
    for (int nt = 0; nt < 4; ++nt) {
      const int nrow = n64 * 64 + nt * 16 + col;  // row of W == column of W^T
      v16h bfr = frag_from_f16row(W + (size_t)nrow * Dv + k0, lane);
      acc[0][nt] = wmma_f16(a0, bfr, acc[0][nt]);
      acc[1][nt] = wmma_f16(a1, bfr, acc[1][nt]);
    }
  }

  // C layout: VGPR r -> row (r + 8*(lane>=16)); column = lane&15.
  const int rhalf = (lane >> 4) << 3;
#pragma unroll
  for (int half = 0; half < 2; ++half) {
#pragma unroll
    for (int nt = 0; nt < 4; ++nt) {
#pragma unroll
      for (int r = 0; r < 8; ++r) {
        const int mm = mtile * 32 + half * 16 + rhalf + r;
        const int n  = n64 * 64 + nt * 16 + col;
        const float val = acc[half][nt][r] + bias[n];
        const int bb = mm >> 11;        // / S
        const int ss = mm & (Sv - 1);
        const int hh = n >> 6;          // / DK
        const int dk = n & (DKv - 1);
        Yh[(((size_t)(bb * Hv + hh)) * Sv + ss) * DKv + dk] = (_Float16)val;
      }
    }
  }
}

// ---------------------------------------------------------------------------
// Kernel 2: causal flash attention.  One wave per (b,h,16-query tile),
// 4 waves per block.  K/V 32x64 tiles double-buffered in LDS via async
// loads: while block i is computed, block i+1 streams in (ASYNCcnt<=16 wait).
// ---------------------------------------------------------------------------
__global__ void __launch_bounds__(128)
attn_kernel(const _Float16* __restrict__ Qh, const _Float16* __restrict__ Kh,
            const _Float16* __restrict__ Vh, _Float16* __restrict__ Oh) {
  __shared__ __align__(16) _Float16 KT[4][2][32][DKv];  // 8 KiB per wave
  __shared__ __align__(16) _Float16 VT[4][2][32][DKv];  // 8 KiB per wave
  __shared__ __align__(16) _Float16 Plds[4][16][32];    // 1 KiB per wave

  const int lane = threadIdx.x & 31;
  const int wid  = threadIdx.x >> 5;
  const int gw   = blockIdx.x * 4 + wid;
  const int QT   = Sv / 16;            // 128 query tiles per (b,h)
  const int qt   = gw % QT;
  const int bh   = gw / QT;
  const int b    = bh / Hv;
  const int h    = bh % Hv;
  const int qbase = qt * 16;

  const _Float16* Qb = Qh + (size_t)(b * Hv + h) * Sv * DKv;
  const _Float16* Kb = Kh + (size_t)(b * Hv + h) * Sv * DKv;
  const _Float16* Vb = Vh + (size_t)(b * Hv + h) * Sv * DKv;

  const int col   = lane & 15;
  const int rhalf = (lane >> 4) << 3;
  const int kbase = (lane >> 4) << 3;

  // Q tile as two A fragments (head dim split into 2 K=32 chunks);
  // fold 1/sqrt(DK)=0.125 (exact in f16) into Q.
  const _Float16 qs = (_Float16)0.125f;
  const int qrow = qbase + col;
  v16h aq0 = frag_from_f16row_scaled(Qb + (size_t)qrow * DKv + 0,  lane, qs);
  v16h aq1 = frag_from_f16row_scaled(Qb + (size_t)qrow * DKv + 32, lane, qs);

  float m[8], l[8];
#pragma unroll
  for (int r = 0; r < 8; ++r) { m[r] = -3.0e38f; l[r] = 0.0f; }
  const v8f z = {0.f, 0.f, 0.f, 0.f, 0.f, 0.f, 0.f, 0.f};
  v8f o[4] = {z, z, z, z};

  const int kend = qbase + 16;           // causal: keys 0 .. qbase+15

  // Prologue: stage block 0 into buffer 0 (16 async b128 issues).
  stage_tile_to_lds(Kb, &KT[wid][0][0][0], lane);
  stage_tile_to_lds(Vb, &VT[wid][0][0][0], lane);

  int buf = 0;
  for (int k0 = 0; k0 < kend; k0 += 32) {
    const bool hasnext = (k0 + 32 < kend);
    if (hasnext) {
      // Prior DS reads of the other buffer drained long ago (consumed by
      // WMMA), but ASYNCcnt/DScnt are unordered -> explicit fence.
      __asm__ volatile("s_wait_dscnt 0" ::: "memory");
      __builtin_amdgcn_wave_barrier();
      stage_tile_to_lds(Kb + (size_t)(k0 + 32) * DKv, &KT[wid][buf ^ 1][0][0], lane);
      stage_tile_to_lds(Vb + (size_t)(k0 + 32) * DKv, &VT[wid][buf ^ 1][0][0], lane);
    }
#if USE_ASYNC_LDS
    // Async loads complete in order: <=16 outstanding means the 16 issues of
    // the *current* buffer have landed; the next buffer keeps streaming.
    if (hasnext) __builtin_amdgcn_s_wait_asynccnt(16);
    else         __builtin_amdgcn_s_wait_asynccnt(0);
#else
    __asm__ volatile("s_wait_dscnt 0" ::: "memory");
#endif
    __builtin_amdgcn_wave_barrier();

    // ---- scores: two 16-key tiles --------------------------------------
    v8f s0 = z, s1 = z;
    {
      v16h bk = frag_from_f16row(&KT[wid][buf][col][0], lane);
      s0 = wmma_f16(aq0, bk, s0);
      bk = frag_from_f16row(&KT[wid][buf][col][32], lane);
      s0 = wmma_f16(aq1, bk, s0);
      bk = frag_from_f16row(&KT[wid][buf][16 + col][0], lane);
      s1 = wmma_f16(aq0, bk, s1);
      bk = frag_from_f16row(&KT[wid][buf][16 + col][32], lane);
      s1 = wmma_f16(aq1, bk, s1);
    }

    // ---- online softmax (per-row over 16-lane halves) ------------------
    const bool needmask = (k0 + 31 > qbase);
    float p0a[8], p1a[8];
#pragma unroll
    for (int r = 0; r < 8; ++r) {
      float sa = s0[r], sb = s1[r];
      if (needmask) {
        const int qq = qbase + rhalf + r;
        if (k0 + col > qq)      sa = -1.0e30f;
        if (k0 + 16 + col > qq) sb = -1.0e30f;
      }
      float v = fmaxf(sa, sb);
      v = fmaxf(v, __shfl_xor(v, 1, 32));
      v = fmaxf(v, __shfl_xor(v, 2, 32));
      v = fmaxf(v, __shfl_xor(v, 4, 32));
      v = fmaxf(v, __shfl_xor(v, 8, 32));
      const float mn = fmaxf(m[r], v);
      const float sc = __expf(m[r] - mn);
      const float pa = __expf(sa - mn);
      const float pb = __expf(sb - mn);
      float rs = pa + pb;
      rs += __shfl_xor(rs, 1, 32);
      rs += __shfl_xor(rs, 2, 32);
      rs += __shfl_xor(rs, 4, 32);
      rs += __shfl_xor(rs, 8, 32);
      l[r] = l[r] * sc + rs;
      m[r] = mn;
      o[0][r] *= sc; o[1][r] *= sc; o[2][r] *= sc; o[3][r] *= sc;
      p0a[r] = pa; p1a[r] = pb;
    }

    // ---- transpose P (C layout -> A layout) through per-wave LDS -------
#pragma unroll
    for (int r = 0; r < 8; ++r) {
      Plds[wid][rhalf + r][col]      = (_Float16)p0a[r];
      Plds[wid][rhalf + r][16 + col] = (_Float16)p1a[r];
    }
    __builtin_amdgcn_wave_barrier();
    __asm__ volatile("s_wait_dscnt 0" ::: "memory");   // same-wave DS RAW
    v16h ap = frag_from_f16row(&Plds[wid][col][0], lane);
    __builtin_amdgcn_wave_barrier();

    // ---- o += P @ V (V fragments gathered from LDS tile) ---------------
#pragma unroll
    for (int dt = 0; dt < 4; ++dt) {
      v16h bv;
#pragma unroll
      for (int i = 0; i < 8; ++i) {
        bv[i]     = VT[wid][buf][kbase + i][dt * 16 + col];
        bv[i + 8] = VT[wid][buf][kbase + 16 + i][dt * 16 + col];
      }
      o[dt] = wmma_f16(ap, bv, o[dt]);
    }

    buf ^= 1;
  }

  // ---- normalize + store head tile into [B,S,D] (head-concat) ----------
#pragma unroll
  for (int dt = 0; dt < 4; ++dt) {
#pragma unroll
    for (int r = 0; r < 8; ++r) {
      const int qq = qbase + rhalf + r;
      const float inv = 1.0f / l[r];
      Oh[((size_t)(b * Sv + qq)) * Dv + h * DKv + dt * 16 + col] =
          (_Float16)(o[dt][r] * inv);
    }
  }
}

// ---------------------------------------------------------------------------
// Kernel 3: output projection.  y = head @ WO^T + b, fp32 out.  32x64 tiles.
// ---------------------------------------------------------------------------
__global__ void __launch_bounds__(256)
out_proj_kernel(const _Float16* __restrict__ Xh, const _Float16* __restrict__ W,
                const float* __restrict__ bias, float* __restrict__ Y) {
  const int lane = threadIdx.x & 31;
  const int gw   = blockIdx.x * 8 + (threadIdx.x >> 5);
  const int NT   = Dv / 64;                 // 12
  const int mtile = gw / NT;
  const int n64   = gw % NT;
  const int col = lane & 15;
  const int m0  = mtile * 32 + col;
  const int m1  = m0 + 16;

  const v8f z = {0.f, 0.f, 0.f, 0.f, 0.f, 0.f, 0.f, 0.f};
  v8f acc[2][4] = {{z, z, z, z}, {z, z, z, z}};

  for (int k0 = 0; k0 < Dv; k0 += 32) {
    v16h a0 = frag_from_f16row(Xh + (size_t)m0 * Dv + k0, lane);
    v16h a1 = frag_from_f16row(Xh + (size_t)m1 * Dv + k0, lane);
#pragma unroll
    for (int nt = 0; nt < 4; ++nt) {
      const int nrow = n64 * 64 + nt * 16 + col;
      v16h bfr = frag_from_f16row(W + (size_t)nrow * Dv + k0, lane);
      acc[0][nt] = wmma_f16(a0, bfr, acc[0][nt]);
      acc[1][nt] = wmma_f16(a1, bfr, acc[1][nt]);
    }
  }

  const int rhalf = (lane >> 4) << 3;
#pragma unroll
  for (int half = 0; half < 2; ++half) {
#pragma unroll
    for (int nt = 0; nt < 4; ++nt) {
#pragma unroll
      for (int r = 0; r < 8; ++r) {
        const int mm = mtile * 32 + half * 16 + rhalf + r;
        const int n  = n64 * 64 + nt * 16 + col;
        Y[(size_t)mm * Dv + n] = acc[half][nt][r] + bias[n];
      }
    }
  }
}

// ---------------------------------------------------------------------------
extern "C" void kernel_launch(void* const* d_in, const int* in_sizes, int n_in,
                              void* d_out, int out_size, void* d_ws, size_t ws_size,
                              hipStream_t stream) {
  (void)in_sizes; (void)n_in; (void)out_size; (void)ws_size;
  const float* Q  = (const float*)d_in[0];
  const float* K  = (const float*)d_in[1];
  const float* V  = (const float*)d_in[2];
  const float* WQ = (const float*)d_in[3];
  const float* bQ = (const float*)d_in[4];
  const float* WK = (const float*)d_in[5];
  const float* bK = (const float*)d_in[6];
  const float* WV = (const float*)d_in[7];
  const float* bV = (const float*)d_in[8];
  const float* WO = (const float*)d_in[9];
  const float* bO = (const float*)d_in[10];
  // d_in[11] = mask: causal, computed analytically in-kernel.

  _Float16* ws = (_Float16*)d_ws;
  const size_t NH = (size_t)Bv * Sv * Dv;   // 3,145,728 halves = 6 MiB
  const size_t WN = (size_t)Dv * Dv;        // 589,824 halves
  _Float16* Qh  = ws;                       // projected heads (f16)
  _Float16* Kh  = ws + NH;
  _Float16* Vh  = ws + 2 * NH;
  _Float16* Oh  = ws + 3 * NH;              // attention output (f16)
  _Float16* WQh = ws + 4 * NH;              // f16 weights
  _Float16* WKh = WQh + WN;
  _Float16* WVh = WKh + WN;
  _Float16* WOh = WVh + WN;
  _Float16* Xq  = WOh + WN;                 // f16 activations
  _Float16* Xk  = Xq + NH;
  _Float16* Xv  = Xk + NH;                  // total ~46.5 MiB of workspace

  const dim3 blk(256);

  // 0) one-time downconverts (bandwidth-trivial: ~71 MB total)
  const int wn4 = (int)(WN / 4);
  const int xn4 = (int)(NH / 4);
  const dim3 wgrid((wn4 + 255) / 256);
  const dim3 xgrid((xn4 + 255) / 256);
  cvt_f32_to_f16_kernel<<<wgrid, blk, 0, stream>>>(WQ, WQh, wn4);
  cvt_f32_to_f16_kernel<<<wgrid, blk, 0, stream>>>(WK, WKh, wn4);
  cvt_f32_to_f16_kernel<<<wgrid, blk, 0, stream>>>(WV, WVh, wn4);
  cvt_f32_to_f16_kernel<<<wgrid, blk, 0, stream>>>(WO, WOh, wn4);
  cvt_f32_to_f16_kernel<<<xgrid, blk, 0, stream>>>(Q, Xq, xn4);
  cvt_f32_to_f16_kernel<<<xgrid, blk, 0, stream>>>(K, Xk, xn4);
  cvt_f32_to_f16_kernel<<<xgrid, blk, 0, stream>>>(V, Xv, xn4);

  // 1) QKV projections: 3 mats x 128 mtiles x 12 ntiles = 4608 waves
  qkv_proj_kernel<<<dim3((3 * 128 * 12) / 8), blk, 0, stream>>>(
      Xq, Xk, Xv, WQh, bQ, WKh, bK, WVh, bV, Qh, Kh, Vh);
  // 2) attention: 2*12*128 = 3072 waves, 4 waves/block
  attn_kernel<<<dim3((Bv * Hv * (Sv / 16)) / 4), dim3(128), 0, stream>>>(
      Qh, Kh, Vh, Oh);
  // 3) output projection: 128 mtiles x 12 ntiles = 1536 waves
  out_proj_kernel<<<dim3((128 * 12) / 8), blk, 0, stream>>>(
      Oh, WOh, bO, (float*)d_out);
}
